// REPLAY_23235773071362
// MI455X (gfx1250) — matrix operand order repaired and checked by reading, not compiled, not run on hardware
//
#include <hip/hip_runtime.h>
#include <stdint.h>

#define S_LEN 256
#define U_DIM 64
#define H_DIM 512
#define H2_DIM 256
#define V_DIM 5000
#define V_PAD 5056                  // 79 * 64
#define NROWS (S_LEN * U_DIM)      // 16384
#define KFC 1280
#define Y_ELEMS ((size_t)NROWS * V_DIM)

#define LDA 20                      // A-tile LDS row stride (dwords): 16B-aligned rows, conflict-free
#define LDB 68                      // B-tile LDS row stride (dwords)

typedef __attribute__((ext_vector_type(16))) __bf16 v16bf;
typedef __attribute__((ext_vector_type(8)))  float  v8f;

union FragAB { unsigned u[8]; v16bf v; };
union FragC  { float    f[8]; v8f   v; };

__device__ __forceinline__ unsigned short f2bf(float f) {
  unsigned x = __builtin_bit_cast(unsigned, f);
  x += 0x7fffu + ((x >> 16) & 1u);
  return (unsigned short)(x >> 16);
}
__device__ __forceinline__ unsigned pack2(float lo, float hi) {
  return (unsigned)f2bf(lo) | ((unsigned)f2bf(hi) << 16);
}

__device__ __forceinline__ void wmma_bf(FragC& c, const FragAB& a, const FragAB& b) {
  c.v = __builtin_amdgcn_wmma_f32_16x16x32_bf16(false, a.v, false, b.v,
                                                (short)0, c.v, false, false);
}

// A fragment: row-major LDS of K-pairs (u32), row stride LDA dwords.
__device__ __forceinline__ void load_fragA(FragAB& a, const unsigned* lds,
                                           int row, int hi) {
#pragma unroll
  for (int i = 0; i < 8; ++i) {
    int kp = (i < 4 ? 0 : 8) + hi * 4 + (i & 3);
    a.u[i] = lds[row * LDA + kp];
  }
}
// B fragment: LDS tile is [16 K-pairs][LDB] u32.
__device__ __forceinline__ void load_fragB(FragAB& b, const unsigned* lds,
                                           int col, int hi) {
#pragma unroll
  for (int i = 0; i < 8; ++i) b.u[i] = lds[((i < 4 ? 0 : 8) + hi * 4 + (i & 3)) * LDB + col];
}

__device__ __forceinline__ void grid_bar(unsigned* cnt, unsigned target) {
  __syncthreads();
  if (threadIdx.x == 0) {
    __threadfence();
    __hip_atomic_fetch_add(cnt, 1u, __ATOMIC_ACQ_REL, __HIP_MEMORY_SCOPE_AGENT);
    while (__hip_atomic_load(cnt, __ATOMIC_ACQUIRE, __HIP_MEMORY_SCOPE_AGENT) < target)
      __builtin_amdgcn_s_sleep(2);
  }
  __syncthreads();
}

// ---------------- prep kernels ----------------
__global__ void k_init(unsigned* bar) { if (threadIdx.x == 0 && blockIdx.x == 0) *bar = 0u; }

__global__ void k_cvt_bf16(const float* src, unsigned short* dst, int n) {
  int i = blockIdx.x * blockDim.x + threadIdx.x;
  if (i < n) dst[i] = f2bf(src[i]);
}

__global__ void k_gather_user(const float* user_w, const int* au, unsigned short* pu) {
  int i = blockIdx.x * blockDim.x + threadIdx.x;
  if (i < U_DIM * H_DIM) {
    int u = i >> 9, j = i & 511;
    pu[i] = f2bf(user_w[(size_t)au[u] * H_DIM + j]);
  }
}

// weight [N][K] row-major f32 -> dst u32 [(K/2)][NPAD] (K-pair packed, zero pad)
__global__ void k_pack_bmat(const float* w, unsigned* dst, int N, int K, int NPAD) {
  int i = blockIdx.x * blockDim.x + threadIdx.x;
  int tot = (K >> 1) * NPAD;
  if (i < tot) {
    int kp = i / NPAD, n = i - kp * NPAD;
    unsigned v = 0u;
    if (n < N) {
      const float* p = w + (size_t)n * K + 2 * kp;
      v = pack2(p[0], p[1]);
    }
    dst[i] = v;
  }
}

// [W_ih ; W_hh] -> u32 [(1024/2)][512]
__global__ void k_pack_wcat(const float* Wih, const float* Whh, unsigned* dst) {
  int i = blockIdx.x * blockDim.x + threadIdx.x;
  if (i < 512 * 512) {
    int kp = i >> 9, n = i & 511;
    int j = kp * 2;
    const float* src = (j < 512) ? (Wih + (size_t)n * 512 + j)
                                 : (Whh + (size_t)n * 512 + (j - 512));
    dst[i] = pack2(src[0], src[1]);
  }
}

// time-embedding table per slot
__global__ void k_week_table(const float* wwi, const float* sigma, const int* wmat,
                             const float* wenc, unsigned short* table) {
  __shared__ float esh[168];
  __shared__ int   wmr[168];
  __shared__ float inv_sum;
  int slot = blockIdx.x, tid = threadIdx.x;
  float sg = fabsf(sigma[slot]);
  float inv2s2 = 1.0f / (2.0f * sg * sg);
  if (tid < 168) {
    float xv = wwi[slot * 168 + tid];
    esh[tid] = expf(-xv * xv * inv2s2);   // row prefactor cancels in normalization
    wmr[tid] = wmat[slot * 168 + tid];
  }
  __syncthreads();
  if (tid == 0) {
    float s = 0.f;
    for (int k = 0; k < 168; ++k) s += esh[k];
    inv_sum = 1.0f / s;
  }
  __syncthreads();
  float acc = 0.f;
  for (int k = 0; k < 168; ++k) acc += esh[k] * wenc[wmr[k] * H2_DIM + tid];
  table[slot * H2_DIM + tid] = f2bf(acc * inv_sum);
}

// fill p_u and t_emb2 sections of out_pu
__global__ void k_fill_apu(const unsigned* pu_u32, const unsigned* table_u32,
                           const int* ytslot, unsigned* apu_u32) {
  int i = blockIdx.x * blockDim.x + threadIdx.x;
  if (i < NROWS * 384) {
    int row = i / 384, c = i - row * 384;
    unsigned v = (c < 256) ? pu_u32[(row & 63) * 256 + c]
                           : table_u32[ytslot[row] * 128 + (c - 256)];
    apu_u32[(size_t)row * 640 + (c + 256)] = v;
  }
}

// ---------------- GEMM 1: poi_time = [x_emb | t_emb1] @ fcpt_w^T + b ----------------
__global__ void k_gemm_pt(const int* x, const int* tslot,
                          const unsigned* enc_u32, const unsigned* table_u32,
                          const unsigned* fcpt_pk, const float* fcptb,
                          unsigned short* poi) {
  __shared__ __attribute__((aligned(16))) unsigned lA[64 * LDA];
  __shared__ __attribute__((aligned(16))) unsigned lB[16 * LDB];
  __shared__ int rowx[64], rowt[64];
  int tid = threadIdx.x;
  int r0 = blockIdx.x * 64, n0 = blockIdx.y * 64;
  if (tid < 64) { rowx[tid] = x[r0 + tid]; rowt[tid] = tslot[r0 + tid]; }
  __syncthreads();
  int lane = tid & 31, wave = tid >> 5;
  int m = lane & 15, hi = lane >> 4, n15 = lane & 15;
  int msub = wave & 3, nsub0 = (wave >> 2) * 2;
  // fill mapping
  int ar = tid >> 2, aq = (tid & 3) * 4;           // A: 64 rows x 16 dw, 4 dw/thread
  int bkp = tid >> 4, bn = (tid & 15) * 4;         // B: 16 rows x 64 dw, 4 dw/thread
  int myx = rowx[ar], myt = rowt[ar];
  const uint4* aEnc = (const uint4*)(enc_u32 + (size_t)myx * 256 + aq);
  const uint4* aTab = (const uint4*)(table_u32 + (size_t)myt * 128 + aq);
  FragC c0, c1;
#pragma unroll
  for (int i = 0; i < 8; ++i) { c0.f[i] = 0.f; c1.f[i] = 0.f; }
  uint4 ra = aEnc[0];
  uint4 rb = *(const uint4*)(fcpt_pk + (size_t)bkp * 512 + n0 + bn);
  for (int kc = 0; kc < 24; ++kc) {
    __syncthreads();
    *(uint4*)&lA[ar * LDA + aq] = ra;
    *(uint4*)&lB[bkp * LDB + bn] = rb;
    __syncthreads();
    if (kc < 23) {
      int kn = kc + 1;
      ra = (kn < 16) ? aEnc[kn * 4] : aTab[(kn - 16) * 4];
      rb = *(const uint4*)(fcpt_pk + (size_t)(kn * 16 + bkp) * 512 + n0 + bn);
    }
    FragAB a, b0, b1;
    load_fragA(a, lA, msub * 16 + m, hi);
    load_fragB(b0, lB, nsub0 * 16 + n15, hi);
    load_fragB(b1, lB, (nsub0 + 1) * 16 + n15, hi);
    wmma_bf(c0, a, b0);
    wmma_bf(c1, a, b1);
  }
#pragma unroll
  for (int rr = 0; rr < 8; ++rr) {
    int gr = r0 + msub * 16 + rr + 8 * hi;
    int g0 = n0 + nsub0 * 16 + n15, g1 = g0 + 16;
    poi[(size_t)gr * H_DIM + g0] = f2bf(c0.f[rr] + fcptb[g0]);
    poi[(size_t)gr * H_DIM + g1] = f2bf(c1.f[rr] + fcptb[g1]);
  }
}

// ---------------- RNN scan (8 persistent blocks + device barrier per step) ----------------
__global__ void k_rnn(const unsigned* poi_u32, unsigned short* hA0, unsigned short* hA1,
                      const unsigned* wcat_pk, const float* bih, const float* bhh,
                      unsigned short* outbf, float* hfinal, unsigned* bar) {
  __shared__ __attribute__((aligned(16))) unsigned lA[64 * LDA];
  __shared__ __attribute__((aligned(16))) unsigned lB[16 * LDB];
  int tid = threadIdx.x;
  int n0 = blockIdx.x * 64;
  int lane = tid & 31, wave = tid >> 5;
  int m = lane & 15, hi = lane >> 4, n15 = lane & 15;
  int msub = wave & 3, nsub0 = (wave >> 2) * 2;
  int ar = tid >> 2, aq = (tid & 3) * 4;
  int bkp = tid >> 4, bn = (tid & 15) * 4;
  unsigned target = 8;
  for (int s = 0; s < S_LEN; ++s) {
    unsigned short* cur = (s & 1) ? hA1 : hA0;
    unsigned short* nxt = (s & 1) ? hA0 : hA1;
    const uint4* aPoi = (const uint4*)(poi_u32 + (size_t)(s * 64 + ar) * 256 + aq);
    const uint4* aH   = (const uint4*)((const unsigned*)cur + (size_t)ar * 256 + aq);
    FragC c0, c1;
#pragma unroll
    for (int i = 0; i < 8; ++i) { c0.f[i] = 0.f; c1.f[i] = 0.f; }
    uint4 ra = aPoi[0];
    uint4 rb = *(const uint4*)(wcat_pk + (size_t)bkp * 512 + n0 + bn);
    for (int kc = 0; kc < 32; ++kc) {
      __syncthreads();
      *(uint4*)&lA[ar * LDA + aq] = ra;
      *(uint4*)&lB[bkp * LDB + bn] = rb;
      __syncthreads();
      if (kc < 31) {
        int kn = kc + 1;
        ra = (kn < 16) ? aPoi[kn * 4] : aH[(kn - 16) * 4];
        rb = *(const uint4*)(wcat_pk + (size_t)(kn * 16 + bkp) * 512 + n0 + bn);
      }
      FragAB a, b0, b1;
      load_fragA(a, lA, msub * 16 + m, hi);
      load_fragB(b0, lB, nsub0 * 16 + n15, hi);
      load_fragB(b1, lB, (nsub0 + 1) * 16 + n15, hi);
      wmma_bf(c0, a, b0);
      wmma_bf(c1, a, b1);
    }
#pragma unroll
    for (int rr = 0; rr < 8; ++rr) {
      int u = msub * 16 + rr + 8 * hi;
      int g0 = n0 + nsub0 * 16 + n15, g1 = g0 + 16;
      float v0 = tanhf(c0.f[rr] + bih[g0] + bhh[g0]);
      float v1 = tanhf(c1.f[rr] + bih[g1] + bhh[g1]);
      nxt[u * H_DIM + g0] = f2bf(v0);
      nxt[u * H_DIM + g1] = f2bf(v1);
      outbf[(size_t)u * S_LEN * H_DIM + (size_t)s * H_DIM + g0] = f2bf(v0);
      outbf[(size_t)u * S_LEN * H_DIM + (size_t)s * H_DIM + g1] = f2bf(v1);
      if (s == S_LEN - 1) { hfinal[u * H_DIM + g0] = v0; hfinal[u * H_DIM + g1] = v1; }
    }
    grid_bar(bar, target);
    target += 8;
  }
}

// ---------------- attention weights (normalized, causal) ----------------
__global__ void k_attw(const float* t, const float* sxy, unsigned short* Wn) {
  __shared__ float red[256];
  int b = blockIdx.x;
  int u = b & (U_DIM - 1), i = b >> 6;
  int j = threadIdx.x;
  int ri = i * U_DIM + u, rj = j * U_DIM + u;
  float dt = t[ri] - t[rj];
  float dx = sxy[ri * 2] - sxy[rj * 2];
  float dy = sxy[ri * 2 + 1] - sxy[rj * 2 + 1];
  const float W0 = 6.283185307179586f / 86400.0f;
  float ft = (cosf(dt * W0) + 1.0f) * 0.5f * expf(-dt * (0.1f / 86400.0f));
  float fs = expf(-100.0f * sqrtf(dx * dx + dy * dy));
  float w = (j <= i) ? (ft * fs + 1e-10f) : 0.0f;
  red[j] = w;
  __syncthreads();
  for (int off = 128; off > 0; off >>= 1) {
    if (j < off) red[j] += red[j + off];
    __syncthreads();
  }
  Wn[((size_t)(u * S_LEN + i) << 8) + j] = f2bf(w / red[0]);
}

// ---------------- GEMM 3: out_w[u] = Wn[u] @ out[u]  (batched over u) ----------------
__global__ void k_gemm_att(const unsigned* Wn_u32, const unsigned short* outbf,
                           unsigned short* apu) {
  __shared__ __attribute__((aligned(16))) unsigned lA[64 * LDA];
  __shared__ __attribute__((aligned(16))) unsigned lB[16 * LDB];
  int bid = blockIdx.x;
  int u = bid >> 5, mt = (bid >> 3) & 3, nt = bid & 7;
  int tid = threadIdx.x;
  int i0 = mt * 64, n0 = nt * 64;
  int lane = tid & 31, wave = tid >> 5;
  int m = lane & 15, hi = lane >> 4, n15 = lane & 15;
  int msub = wave & 3, nsub0 = (wave >> 2) * 2;
  int ar = tid >> 2, aq = (tid & 3) * 4;
  int bkp = tid >> 4, bn = (tid & 15) * 4;
  const unsigned short* outu = outbf + (size_t)u * S_LEN * H_DIM;
  const uint4* aW = (const uint4*)(Wn_u32 + (size_t)(u * S_LEN + i0 + ar) * 128 + aq);
  FragC c0, c1;
#pragma unroll
  for (int i = 0; i < 8; ++i) { c0.f[i] = 0.f; c1.f[i] = 0.f; }
  auto loadB = [&](int kc) -> uint4 {
    const unsigned short* p0 = outu + (size_t)(kc * 32 + bkp * 2) * H_DIM + n0 + bn;
    uint2 lo = *(const uint2*)p0;
    uint2 hb = *(const uint2*)(p0 + H_DIM);
    uint4 r;
    r.x = (lo.x & 0xffffu) | (hb.x << 16);
    r.y = (lo.x >> 16)     | (hb.x & 0xffff0000u);
    r.z = (lo.y & 0xffffu) | (hb.y << 16);
    r.w = (lo.y >> 16)     | (hb.y & 0xffff0000u);
    return r;
  };
  uint4 ra = aW[0];
  uint4 rb = loadB(0);
  for (int kc = 0; kc < 8; ++kc) {
    __syncthreads();
    *(uint4*)&lA[ar * LDA + aq] = ra;
    *(uint4*)&lB[bkp * LDB + bn] = rb;
    __syncthreads();
    if (kc < 7) {
      ra = aW[(kc + 1) * 4];
      rb = loadB(kc + 1);
    }
    FragAB a, b0, b1;
    load_fragA(a, lA, msub * 16 + m, hi);
    load_fragB(b0, lB, nsub0 * 16 + n15, hi);
    load_fragB(b1, lB, (nsub0 + 1) * 16 + n15, hi);
    wmma_bf(c0, a, b0);
    wmma_bf(c1, a, b1);
  }
#pragma unroll
  for (int rr = 0; rr < 8; ++rr) {
    int gi = i0 + msub * 16 + rr + 8 * hi;
    int g0 = n0 + nsub0 * 16 + n15, g1 = g0 + 16;
    apu[(size_t)(gi * U_DIM + u) * KFC + g0] = f2bf(c0.f[rr]);
    apu[(size_t)(gi * U_DIM + u) * KFC + g1] = f2bf(c1.f[rr]);
  }
}

// ---------------- GEMM 4: y = out_pu @ fc_w^T + fc_b  (dominant) ----------------
__global__ void k_gemm_fc(const unsigned* apu_u32, const unsigned* fcw_pk,
                          const float* fcb, float* yout) {
  __shared__ __attribute__((aligned(16))) unsigned lA[128 * LDA];
  __shared__ __attribute__((aligned(16))) unsigned lB[16 * LDB];
  int tid = threadIdx.x;
  int r0 = blockIdx.x * 128, n0 = blockIdx.y * 64;
  int lane = tid & 31, wave = tid >> 5;
  int m = lane & 15, hi = lane >> 4, n15 = lane & 15;
  int ar = tid >> 1, ah = (tid & 1) * 8;           // A: 128 rows x 16 dw, 8 dw/thread
  int bkp = tid >> 4, bn = (tid & 15) * 4;         // B: 16 rows x 64 dw, 4 dw/thread
  const uint4* aRow = (const uint4*)(apu_u32 + (size_t)(r0 + ar) * 640);
  int ai = ah >> 2;                                // uint4 index offset within chunk (0 or 2)
  FragC c[4];
#pragma unroll
  for (int ns = 0; ns < 4; ++ns)
#pragma unroll
    for (int i = 0; i < 8; ++i) c[ns].f[i] = 0.f;
  uint4 ra0 = aRow[ai], ra1 = aRow[ai + 1];
  uint4 rb = *(const uint4*)(fcw_pk + (size_t)bkp * V_PAD + n0 + bn);
  for (int kc = 0; kc < 40; ++kc) {
    __syncthreads();
    *(uint4*)&lA[ar * LDA + ah] = ra0;
    *(uint4*)&lA[ar * LDA + ah + 4] = ra1;
    *(uint4*)&lB[bkp * LDB + bn] = rb;
    __syncthreads();
    if (kc < 39) {
      int kn = kc + 1;
      ra0 = aRow[kn * 4 + ai];
      ra1 = aRow[kn * 4 + ai + 1];
      rb = *(const uint4*)(fcw_pk + (size_t)(kn * 16 + bkp) * V_PAD + n0 + bn);
    }
    if (kc < 38) {
      int kp2 = kc + 2;
      __builtin_prefetch((const void*)(aRow + kp2 * 4 + ai), 0, 3);
      __builtin_prefetch((const void*)(fcw_pk + (size_t)(kp2 * 16 + bkp) * V_PAD + n0 + bn), 0, 3);
    }
    FragAB a;
    load_fragA(a, lA, wave * 16 + m, hi);
#pragma unroll
    for (int ns = 0; ns < 4; ++ns) {
      FragAB b;
      load_fragB(b, lB, ns * 16 + n15, hi);
      wmma_bf(c[ns], a, b);
    }
  }
#pragma unroll
  for (int ns = 0; ns < 4; ++ns) {
    int gn = n0 + ns * 16 + n15;
    if (gn < V_DIM) {
      float bias = fcb[gn];
#pragma unroll
      for (int rr = 0; rr < 8; ++rr) {
        int gr = r0 + wave * 16 + rr + 8 * hi;
        yout[(size_t)gr * V_DIM + gn] = c[ns].f[rr] + bias;
      }
    }
  }
}

// ---------------- host ----------------
extern "C" void kernel_launch(void* const* d_in, const int* in_sizes, int n_in,
                              void* d_out, int out_size, void* d_ws, size_t ws_size,
                              hipStream_t stream) {
  (void)in_sizes; (void)n_in; (void)out_size; (void)ws_size;
  const int*   x      = (const int*)  d_in[0];
  const float* t      = (const float*)d_in[1];
  const int*   tslot  = (const int*)  d_in[2];
  const float* sxy    = (const float*)d_in[3];
  const int*   ytslot = (const int*)  d_in[5];
  const float* h0     = (const float*)d_in[7];
  const int*   auser  = (const int*)  d_in[8];
  const int*   wmat   = (const int*)  d_in[9];
  const float* wwi    = (const float*)d_in[10];
  const float* sigma  = (const float*)d_in[11];
  const float* enc_w  = (const float*)d_in[12];
  const float* user_w = (const float*)d_in[13];
  const float* wenc   = (const float*)d_in[14];
  const float* fcptw  = (const float*)d_in[15];
  const float* fcptb  = (const float*)d_in[16];
  const float* Wih    = (const float*)d_in[17];
  const float* Whh    = (const float*)d_in[18];
  const float* bih    = (const float*)d_in[19];
  const float* bhh    = (const float*)d_in[20];
  const float* fcw    = (const float*)d_in[21];
  const float* fcb    = (const float*)d_in[22];

  char* ws = (char*)d_ws;
  size_t o = 0;
  auto alloc = [&](size_t bytes) -> char* {
    size_t r = (o + 255) & ~(size_t)255;
    o = r + bytes;
    return ws + r;
  };
  unsigned*       bar     = (unsigned*)      alloc(4);
  unsigned short* table   = (unsigned short*)alloc((size_t)168 * H2_DIM * 2);
  unsigned short* encb    = (unsigned short*)alloc((size_t)V_DIM * H_DIM * 2);
  unsigned short* pub     = (unsigned short*)alloc((size_t)U_DIM * H_DIM * 2);
  unsigned short* hA0     = (unsigned short*)alloc((size_t)U_DIM * H_DIM * 2);
  unsigned short* hA1     = (unsigned short*)alloc((size_t)U_DIM * H_DIM * 2);
  unsigned*       wcat_pk = (unsigned*)      alloc((size_t)512 * 512 * 4);
  unsigned*       fcpt_pk = (unsigned*)      alloc((size_t)384 * 512 * 4);
  unsigned*       fcw_pk  = (unsigned*)      alloc((size_t)640 * V_PAD * 4);
  unsigned short* poi     = (unsigned short*)alloc((size_t)NROWS * H_DIM * 2);
  unsigned short* outb    = (unsigned short*)alloc((size_t)U_DIM * S_LEN * H_DIM * 2);
  unsigned short* Wn      = (unsigned short*)alloc((size_t)U_DIM * S_LEN * S_LEN * 2);
  unsigned short* apu     = (unsigned short*)alloc((size_t)NROWS * KFC * 2);

  float* yout   = (float*)d_out;
  float* hfinal = yout + Y_ELEMS;

  k_init<<<1, 64, 0, stream>>>(bar);
  k_week_table<<<168, 256, 0, stream>>>(wwi, sigma, wmat, wenc, table);
  k_cvt_bf16<<<(V_DIM * H_DIM + 255) / 256, 256, 0, stream>>>(enc_w, encb, V_DIM * H_DIM);
  k_cvt_bf16<<<(U_DIM * H_DIM + 255) / 256, 256, 0, stream>>>(h0, hA0, U_DIM * H_DIM);
  k_gather_user<<<(U_DIM * H_DIM + 255) / 256, 256, 0, stream>>>(user_w, auser, pub);
  k_pack_wcat<<<(512 * 512 + 255) / 256, 256, 0, stream>>>(Wih, Whh, wcat_pk);
  k_pack_bmat<<<(384 * 512 + 255) / 256, 256, 0, stream>>>(fcptw, fcpt_pk, 512, 768, 512);
  k_pack_bmat<<<(640 * V_PAD + 255) / 256, 256, 0, stream>>>(fcw, fcw_pk, V_DIM, KFC, V_PAD);
  k_fill_apu<<<(NROWS * 384 + 255) / 256, 256, 0, stream>>>(
      (const unsigned*)pub, (const unsigned*)table, ytslot, (unsigned*)apu);

  dim3 gpt(NROWS / 64, 8);
  k_gemm_pt<<<gpt, 256, 0, stream>>>(x, tslot, (const unsigned*)encb,
                                     (const unsigned*)table, fcpt_pk, fcptb, poi);

  k_rnn<<<8, 256, 0, stream>>>((const unsigned*)poi, hA0, hA1, wcat_pk,
                               bih, bhh, outb, hfinal, bar);

  k_attw<<<NROWS, 256, 0, stream>>>(t, sxy, Wn);
  k_gemm_att<<<U_DIM * 32, 256, 0, stream>>>((const unsigned*)Wn, outb, apu);

  dim3 gfc(NROWS / 128, (V_DIM + 63) / 64);
  k_gemm_fc<<<gfc, 256, 0, stream>>>((const unsigned*)apu, fcw_pk, fcb, yout);
}
